// TrackFormerDecoderLayer_39041252721030
// MI455X (gfx1250) — compile-verified, matrix operand order checked by reading, not run on hardware
//
#include <hip/hip_runtime.h>
#include <hip/hip_bf16.h>

typedef __attribute__((ext_vector_type(16))) _Float16 v16h;
typedef __attribute__((ext_vector_type(8)))  _Float16 v8h;
typedef __attribute__((ext_vector_type(4)))  _Float16 v4h;
typedef __attribute__((ext_vector_type(8)))  float    v8f;

#define NB    8
#define QLEN  900
#define ED    256
#define NHD   8
#define HD    32
#define NPTS  4
#define DFF   512
#define HBEV  200
#define WBEV  200
#define LBEV  (HBEV * WBEV)

static __device__ inline v16h cat8(v8h lo, v8h hi) {
    return __builtin_shufflevector(lo, hi, 0, 1, 2, 3, 4, 5, 6, 7,
                                   8, 9, 10, 11, 12, 13, 14, 15);
}

static __device__ inline v8h cvt8(float4 a, float4 b, float s) {
    v8h r;
    r[0] = (_Float16)(a.x * s); r[1] = (_Float16)(a.y * s);
    r[2] = (_Float16)(a.z * s); r[3] = (_Float16)(a.w * s);
    r[4] = (_Float16)(b.x * s); r[5] = (_Float16)(b.y * s);
    r[6] = (_Float16)(b.z * s); r[7] = (_Float16)(b.w * s);
    return r;
}

// ---------------------------------------------------------------------------
// Elementwise helpers
// ---------------------------------------------------------------------------
__global__ void f32_to_f16_kernel(const float* __restrict__ in,
                                  _Float16* __restrict__ out, int n) {
    int i = blockIdx.x * blockDim.x + threadIdx.x;
    if (i < n) out[i] = (_Float16)in[i];
}

__global__ void add_kernel(const float* __restrict__ a,
                           const float* __restrict__ b,
                           float* __restrict__ c, int n4) {
    int i = blockIdx.x * blockDim.x + threadIdx.x;
    if (i < n4) {
        float4 x = ((const float4*)a)[i];
        float4 y = ((const float4*)b)[i];
        float4 z{x.x + y.x, x.y + y.y, x.z + y.z, x.w + y.w};
        ((float4*)c)[i] = z;
    }
}

// out = LayerNorm(a + b) over last dim (ED=256), one block (256 thr) per row
__global__ __launch_bounds__(256) void add_ln_kernel(
    const float* __restrict__ a, const float* __restrict__ b,
    const float* __restrict__ gamma, const float* __restrict__ beta,
    float* __restrict__ out) {
    int row = blockIdx.x;
    int t = threadIdx.x;
    size_t o = (size_t)row * ED + t;
    float x = a[o] + b[o];
    __shared__ float red[8];
    float s = x;
#pragma unroll
    for (int m = 16; m >= 1; m >>= 1) s += __shfl_xor(s, m, 32);
    if ((t & 31) == 0) red[t >> 5] = s;
    __syncthreads();
    float tot = 0.f;
#pragma unroll
    for (int w = 0; w < 8; ++w) tot += red[w];
    float mean = tot * (1.0f / ED);
    __syncthreads();
    float d = x - mean;
    float v = d * d;
#pragma unroll
    for (int m = 16; m >= 1; m >>= 1) v += __shfl_xor(v, m, 32);
    if ((t & 31) == 0) red[t >> 5] = v;
    __syncthreads();
    float vt = 0.f;
#pragma unroll
    for (int w = 0; w < 8; ++w) vt += red[w];
    float var = vt * (1.0f / ED);
    out[o] = d * rsqrtf(var + 1e-5f) * gamma[t] + beta[t];
}

// ---------------------------------------------------------------------------
// WMMA GEMM: C[M,N] = act(A[M,K](f32) * Bw[K,N](f16) + bias[N])
// 256 threads = 8 waves; tile BM=128 x BN=64, BK=32 (K % 32 == 0).
// Double-buffered LDS + register prefetch; one barrier per k-step.
// All fragment ds_load_b128s issue before the 4 back-to-back WMMAs.
// ---------------------------------------------------------------------------
#define BM 128
#define BN 64
#define BK 32

__global__ __launch_bounds__(256) void gemm_f16w_kernel(
    const float* __restrict__ A, const _Float16* __restrict__ Bw,
    const float* __restrict__ bias, float* __restrict__ C,
    int M, int N, int K, int relu) {
    __shared__ _Float16 As[2][BM][BK + 8];   // [buf][row][k]
    __shared__ _Float16 Bt[2][BN][BK + 8];   // [buf][col][k] (transposed)

    int tid = threadIdx.x;
    int wave = tid >> 5;
    int lane = tid & 31;
    int g = lane >> 4;
    int l15 = lane & 15;
    int bm = blockIdx.y * BM;
    int bn = blockIdx.x * BN;

    const bool mfull = (bm + BM <= M);
    const bool nfull = (bn + BN <= N);

    float4 aReg[4];
    v8h bReg;

    auto loadA = [&](int k0) {
        if (mfull) {
#pragma unroll
            for (int i = 0; i < 4; ++i) {
                int e4 = i * 256 + tid;
                int r = e4 >> 3;
                int c4 = (e4 & 7) << 2;
                aReg[i] = *(const float4*)(A + (size_t)(bm + r) * K + k0 + c4);
            }
        } else {
#pragma unroll
            for (int i = 0; i < 4; ++i) {
                int e4 = i * 256 + tid;
                int r = e4 >> 3;
                int c4 = (e4 & 7) << 2;
                int gr = bm + r;
                aReg[i] = (gr < M)
                    ? *(const float4*)(A + (size_t)gr * K + k0 + c4)
                    : float4{0.f, 0.f, 0.f, 0.f};
            }
        }
    };
    auto loadB = [&](int k0) {
        int r = tid >> 3;            // k row 0..31
        int c8 = (tid & 7) << 3;     // col group
        int gc = bn + c8;
        if (nfull) {
            bReg = *(const v8h*)(Bw + (size_t)(k0 + r) * N + gc);
        } else {
#pragma unroll
            for (int j = 0; j < 8; ++j)
                bReg[j] = (gc + j < N) ? Bw[(size_t)(k0 + r) * N + gc + j]
                                       : (_Float16)0.f;
        }
    };
    auto storeTiles = [&](int buf) {
#pragma unroll
        for (int i = 0; i < 4; ++i) {
            int e4 = i * 256 + tid;
            int r = e4 >> 3;
            int c4 = (e4 & 7) << 2;
            v4h hv;
            hv[0] = (_Float16)aReg[i].x; hv[1] = (_Float16)aReg[i].y;
            hv[2] = (_Float16)aReg[i].z; hv[3] = (_Float16)aReg[i].w;
            *(v4h*)(&As[buf][r][c4]) = hv;
        }
        int r = tid >> 3;
        int c8 = (tid & 7) << 3;
#pragma unroll
        for (int j = 0; j < 8; ++j) Bt[buf][c8 + j][r] = bReg[j];
    };

    v8f acc[4] = {};

    loadA(0);
    loadB(0);
    storeTiles(0);
    __syncthreads();

    int cur = 0;
    for (int k0 = 0; k0 < K; k0 += BK) {
        int nxt = k0 + BK;
        if (nxt < K) {               // prefetch next tile into registers
            loadA(nxt);
            loadB(nxt);
        }
        // Issue all fragment LDS loads up front, then 4 back-to-back WMMAs.
        int arow = wave * 16 + l15;
        v16h af = cat8(*(const v8h*)(&As[cur][arow][g << 3]),
                       *(const v8h*)(&As[cur][arow][16 + (g << 3)]));
        v16h bf[4];
#pragma unroll
        for (int t = 0; t < 4; ++t) {
            int bcol = t * 16 + l15;
            bf[t] = cat8(*(const v8h*)(&Bt[cur][bcol][g << 3]),
                         *(const v8h*)(&Bt[cur][bcol][16 + (g << 3)]));
        }
#pragma unroll
        for (int t = 0; t < 4; ++t) {
            acc[t] = __builtin_amdgcn_wmma_f32_16x16x32_f16(
                false, af, false, bf[t], (short)0, acc[t], false, false);
        }
        if (nxt < K) {               // fill the other buffer, single barrier
            storeTiles(cur ^ 1);
            __syncthreads();
            cur ^= 1;
        }
    }

    // Epilogue: bias (+ optional ReLU); C/D layout: VGPR r -> row r + g*8
#pragma unroll
    for (int t = 0; t < 4; ++t) {
        int col = bn + t * 16 + l15;
        float bv = (nfull || col < N) ? bias[col] : 0.f;
#pragma unroll
        for (int r = 0; r < 8; ++r) {
            int row = bm + wave * 16 + r + (g << 3);
            if ((mfull || row < M) && (nfull || col < N)) {
                float v = acc[t][r] + bv;
                if (relu) v = fmaxf(v, 0.f);
                C[(size_t)row * N + col] = v;
            }
        }
    }
}

// ---------------------------------------------------------------------------
// MHA flash-attention: one wave per (batch n, head h, 16-query tile).
// Q/K/V: (NB, S, ED) row-major with head slice [h*HD, h*HD+32).
// Full chunks take the guard-free path; only the tail chunk masks keys.
// ---------------------------------------------------------------------------
template <bool TAIL>
static __device__ inline void attn_step(
    int kk0, int S, int lane, int g, int l15, size_t base,
    const float* __restrict__ Kp, const float* __restrict__ Vp,
    const v16h& qf, v8f& o0, v8f& o1, float* mrow, float* lrow,
    _Float16 (*Pl)[40], _Float16 (*Vl)[40]) {
    const float4 z4{0.f, 0.f, 0.f, 0.f};

    int key0 = kk0 + l15;
    int key1 = kk0 + 16 + l15;
    const float* kp0 = Kp + base + (size_t)key0 * ED;
    const float* kp1 = Kp + base + (size_t)key1 * ED;
    v16h kb0, kb1;
    if (!TAIL || key0 < S) {
        kb0 = cat8(cvt8(*(const float4*)(kp0 + (g << 3)),
                        *(const float4*)(kp0 + (g << 3) + 4), 1.f),
                   cvt8(*(const float4*)(kp0 + 16 + (g << 3)),
                        *(const float4*)(kp0 + 16 + (g << 3) + 4), 1.f));
    } else {
        kb0 = cat8(cvt8(z4, z4, 1.f), cvt8(z4, z4, 1.f));
    }
    if (!TAIL || key1 < S) {
        kb1 = cat8(cvt8(*(const float4*)(kp1 + (g << 3)),
                        *(const float4*)(kp1 + (g << 3) + 4), 1.f),
                   cvt8(*(const float4*)(kp1 + 16 + (g << 3)),
                        *(const float4*)(kp1 + 16 + (g << 3) + 4), 1.f));
    } else {
        kb1 = cat8(cvt8(z4, z4, 1.f), cvt8(z4, z4, 1.f));
    }

    v8f z = {};
    v8f s0 = __builtin_amdgcn_wmma_f32_16x16x32_f16(
        false, qf, false, kb0, (short)0, z, false, false);
    v8f s1 = __builtin_amdgcn_wmma_f32_16x16x32_f16(
        false, qf, false, kb1, (short)0, z, false, false);

    if (TAIL) {
        if (key0 >= S) {
#pragma unroll
            for (int r = 0; r < 8; ++r) s0[r] = -1e30f;
        }
        if (key1 >= S) {
#pragma unroll
            for (int r = 0; r < 8; ++r) s1[r] = -1e30f;
        }
    }

    // Online softmax (row reductions across the 16-lane half-groups)
    float p0[8], p1[8];
#pragma unroll
    for (int r = 0; r < 8; ++r) {
        float cm = fmaxf(s0[r], s1[r]);
        for (int m = 8; m >= 1; m >>= 1)
            cm = fmaxf(cm, __shfl_xor(cm, m, 32));
        float nm = fmaxf(mrow[r], cm);
        float corr = __expf(mrow[r] - nm);
        p0[r] = __expf(s0[r] - nm);
        p1[r] = __expf(s1[r] - nm);
        float cs = p0[r] + p1[r];
        for (int m = 8; m >= 1; m >>= 1) cs += __shfl_xor(cs, m, 32);
        lrow[r] = lrow[r] * corr + cs;
        mrow[r] = nm;
        o0[r] *= corr;
        o1[r] *= corr;
    }

    // P (C-layout) -> LDS [row][key]
#pragma unroll
    for (int r = 0; r < 8; ++r) {
        Pl[r + (g << 3)][l15] = (_Float16)p0[r];
        Pl[r + (g << 3)][16 + l15] = (_Float16)p1[r];
    }
    // V chunk -> LDS transposed [d][key]; lane = key
    {
        int key = kk0 + lane;
        const float* vp = Vp + base + (size_t)key * ED;
#pragma unroll
        for (int j = 0; j < 8; ++j) {
            float4 vv = (!TAIL || key < S) ? *(const float4*)(vp + (j << 2)) : z4;
            Vl[(j << 2) + 0][lane] = (_Float16)vv.x;
            Vl[(j << 2) + 1][lane] = (_Float16)vv.y;
            Vl[(j << 2) + 2][lane] = (_Float16)vv.z;
            Vl[(j << 2) + 3][lane] = (_Float16)vv.w;
        }
    }
    __syncthreads();

    v16h pf = cat8(*(const v8h*)(&Pl[l15][g << 3]),
                   *(const v8h*)(&Pl[l15][16 + (g << 3)]));
    v16h vb0 = cat8(*(const v8h*)(&Vl[l15][g << 3]),
                    *(const v8h*)(&Vl[l15][16 + (g << 3)]));
    v16h vb1 = cat8(*(const v8h*)(&Vl[16 + l15][g << 3]),
                    *(const v8h*)(&Vl[16 + l15][16 + (g << 3)]));
    __syncthreads();

    o0 = __builtin_amdgcn_wmma_f32_16x16x32_f16(
        false, pf, false, vb0, (short)0, o0, false, false);
    o1 = __builtin_amdgcn_wmma_f32_16x16x32_f16(
        false, pf, false, vb1, (short)0, o1, false, false);
}

__global__ __launch_bounds__(32) void mha_attn_kernel(
    const float* __restrict__ Qp, const float* __restrict__ Kp,
    const float* __restrict__ Vp, float* __restrict__ Op, int S) {
    int lane = threadIdx.x;
    int qt = blockIdx.x;
    int h = blockIdx.y;
    int n = blockIdx.z;
    int g = lane >> 4;
    int l15 = lane & 15;
    const float scale = 0.17677669529663687f;  // 1/sqrt(32)

    size_t base = (size_t)n * S * ED + (size_t)h * HD;

    int qrow = qt * 16 + l15;
    if (qrow >= S) qrow = S - 1;
    const float* qp = Qp + base + (size_t)qrow * ED;
    v16h qf = cat8(cvt8(*(const float4*)(qp + (g << 3)),
                        *(const float4*)(qp + (g << 3) + 4), scale),
                   cvt8(*(const float4*)(qp + 16 + (g << 3)),
                        *(const float4*)(qp + 16 + (g << 3) + 4), scale));

    v8f o0 = {}, o1 = {};
    float mrow[8], lrow[8];
#pragma unroll
    for (int r = 0; r < 8; ++r) { mrow[r] = -3.0e38f; lrow[r] = 0.f; }

    __shared__ _Float16 Pl[16][40];
    __shared__ _Float16 Vl[32][40];

    int SFull = S & ~31;
    for (int kk0 = 0; kk0 < SFull; kk0 += 32)
        attn_step<false>(kk0, S, lane, g, l15, base, Kp, Vp, qf,
                         o0, o1, mrow, lrow, Pl, Vl);
    if (SFull < S)
        attn_step<true>(SFull, S, lane, g, l15, base, Kp, Vp, qf,
                        o0, o1, mrow, lrow, Pl, Vl);

#pragma unroll
    for (int r = 0; r < 8; ++r) {
        int row = qt * 16 + r + (g << 3);
        if (row < S) {
            float inv = 1.f / lrow[r];
            Op[base + (size_t)row * ED + l15] = o0[r] * inv;
            Op[base + (size_t)row * ED + 16 + l15] = o1[r] * inv;
        }
    }
}

// ---------------------------------------------------------------------------
// Deformable sampling: one wave per (n, q, h); lane = channel within head.
// ---------------------------------------------------------------------------
__global__ __launch_bounds__(32) void deform_sample_kernel(
    const float* __restrict__ val, const float* __restrict__ ref,
    const float* __restrict__ off, const float* __restrict__ attnl,
    float* __restrict__ outp) {
    int lane = threadIdx.x;
    int q = blockIdx.x;
    int h = blockIdx.y;
    int n = blockIdx.z;
    size_t rq = (size_t)n * QLEN + q;

    float rx = ref[rq * 2 + 0];
    float ry = ref[rq * 2 + 1];

    float a[NPTS];
    float amax = -3e38f;
#pragma unroll
    for (int p = 0; p < NPTS; ++p) {
        a[p] = attnl[rq * (NHD * NPTS) + h * NPTS + p];
        amax = fmaxf(amax, a[p]);
    }
    float asum = 0.f;
#pragma unroll
    for (int p = 0; p < NPTS; ++p) { a[p] = __expf(a[p] - amax); asum += a[p]; }
    float ainv = 1.f / asum;

    const float* vb = val + (size_t)n * LBEV * ED + (size_t)h * HD + lane;
    float acc = 0.f;
#pragma unroll
    for (int p = 0; p < NPTS; ++p) {
        float ox = off[rq * (NHD * NPTS * 2) + (h * NPTS + p) * 2 + 0];
        float oy = off[rq * (NHD * NPTS * 2) + (h * NPTS + p) * 2 + 1];
        float x = (rx + ox * (1.0f / WBEV)) * WBEV - 0.5f;
        float y = (ry + oy * (1.0f / HBEV)) * HBEV - 0.5f;
        float x0f = floorf(x), y0f = floorf(y);
        float fx = x - x0f, fy = y - y0f;
        int x0 = (int)x0f, y0 = (int)y0f;
        float s = 0.f;
#pragma unroll
        for (int c = 0; c < 4; ++c) {
            int dx = c & 1, dy = c >> 1;
            int xi = x0 + dx, yi = y0 + dy;
            float w = fabsf((float)(1 - dx) - fx) * fabsf((float)(1 - dy) - fy);
            if (xi >= 0 && xi < WBEV && yi >= 0 && yi < HBEV) {
                s += w * vb[(size_t)(yi * WBEV + xi) * ED];
            }
        }
        acc += a[p] * ainv * s;
    }
    outp[rq * ED + (size_t)h * HD + lane] = acc;
}

// ---------------------------------------------------------------------------
// Launcher
// ---------------------------------------------------------------------------
extern "C" void kernel_launch(void* const* d_in, const int* in_sizes, int n_in,
                              void* d_out, int out_size, void* d_ws,
                              size_t ws_size, hipStream_t stream) {
    (void)in_sizes; (void)n_in; (void)out_size; (void)ws_size;

    const float* queries = (const float*)d_in[0];
    const float* bev     = (const float*)d_in[1];
    const float* refpts  = (const float*)d_in[2];
    const float* origq   = (const float*)d_in[3];
    const float* Wq = (const float*)d_in[4];  const float* bq = (const float*)d_in[5];
    const float* Wk = (const float*)d_in[6];  const float* bk = (const float*)d_in[7];
    const float* Wv = (const float*)d_in[8];  const float* bv = (const float*)d_in[9];
    const float* Wo = (const float*)d_in[10]; const float* bo = (const float*)d_in[11];
    const float* ln1_g = (const float*)d_in[12]; const float* ln1_b = (const float*)d_in[13];
    const float* dv_W = (const float*)d_in[14];  const float* dv_b = (const float*)d_in[15];
    const float* doff_W = (const float*)d_in[16]; const float* doff_b = (const float*)d_in[17];
    const float* dattn_W = (const float*)d_in[18]; const float* dattn_b = (const float*)d_in[19];
    const float* dout_W = (const float*)d_in[20]; const float* dout_b = (const float*)d_in[21];
    const float* ln2_g = (const float*)d_in[22]; const float* ln2_b = (const float*)d_in[23];
    const float* W1 = (const float*)d_in[24]; const float* b1 = (const float*)d_in[25];
    const float* W2 = (const float*)d_in[26]; const float* b2 = (const float*)d_in[27];
    const float* ln3_g = (const float*)d_in[28]; const float* ln3_b = (const float*)d_in[29];

    float* out = (float*)d_out;

    const int M1 = NB * QLEN;        // 7200
    const int MV = NB * LBEV;        // 320000

    // ---- carve workspace ----
    char* cur = (char*)d_ws;
    auto carve = [&](size_t bytes) -> char* {
        char* p = cur;
        cur += (bytes + 255) & ~(size_t)255;
        return p;
    };
    _Float16* hWq   = (_Float16*)carve((size_t)ED * ED * 2);
    _Float16* hWk   = (_Float16*)carve((size_t)ED * ED * 2);
    _Float16* hWv   = (_Float16*)carve((size_t)ED * ED * 2);
    _Float16* hWo   = (_Float16*)carve((size_t)ED * ED * 2);
    _Float16* hdv   = (_Float16*)carve((size_t)ED * ED * 2);
    _Float16* hdoff = (_Float16*)carve((size_t)ED * 64 * 2);
    _Float16* hdatt = (_Float16*)carve((size_t)ED * 32 * 2);
    _Float16* hdout = (_Float16*)carve((size_t)ED * ED * 2);
    _Float16* hW1   = (_Float16*)carve((size_t)ED * DFF * 2);
    _Float16* hW2   = (_Float16*)carve((size_t)DFF * ED * 2);

    float* qk    = (float*)carve((size_t)M1 * ED * 4);
    float* Qt    = (float*)carve((size_t)M1 * ED * 4);
    float* Kt    = (float*)carve((size_t)M1 * ED * 4);
    float* Vt    = (float*)carve((size_t)M1 * ED * 4);
    float* mo    = (float*)carve((size_t)M1 * ED * 4);
    float* out1  = (float*)carve((size_t)M1 * ED * 4);
    float* out2  = (float*)carve((size_t)M1 * ED * 4);
    float* dq    = (float*)carve((size_t)M1 * ED * 4);
    float* offb  = (float*)carve((size_t)M1 * 64 * 4);
    float* attnb = (float*)carve((size_t)M1 * 32 * 4);
    float* samp  = (float*)carve((size_t)M1 * ED * 4);
    float* dob   = (float*)carve((size_t)M1 * ED * 4);
    float* out3  = (float*)carve((size_t)M1 * ED * 4);
    float* ffh   = (float*)carve((size_t)M1 * DFF * 4);
    float* out4  = (float*)carve((size_t)M1 * ED * 4);
    float* value = (float*)carve((size_t)MV * ED * 4);

    auto cvt = [&](const float* w, _Float16* dst, int n) {
        f32_to_f16_kernel<<<(n + 255) / 256, 256, 0, stream>>>(w, dst, n);
    };
    auto gemm = [&](const float* A, const _Float16* Bh, const float* bias,
                    float* C, int M, int N, int K, int relu) {
        dim3 grid((N + BN - 1) / BN, (M + BM - 1) / BM);
        gemm_f16w_kernel<<<grid, 256, 0, stream>>>(A, Bh, bias, C, M, N, K, relu);
    };

    // ---- weight conversions ----
    cvt(Wq, hWq, ED * ED);   cvt(Wk, hWk, ED * ED);
    cvt(Wv, hWv, ED * ED);   cvt(Wo, hWo, ED * ED);
    cvt(dv_W, hdv, ED * ED); cvt(doff_W, hdoff, ED * 64);
    cvt(dattn_W, hdatt, ED * 32); cvt(dout_W, hdout, ED * ED);
    cvt(W1, hW1, ED * DFF);  cvt(W2, hW2, DFF * ED);

    // ---- MHA block ----
    add_kernel<<<(M1 * ED / 4 + 255) / 256, 256, 0, stream>>>(queries, origq, qk, M1 * ED / 4);
    gemm(qk, hWq, bq, Qt, M1, ED, ED, 0);
    gemm(qk, hWk, bk, Kt, M1, ED, ED, 0);
    gemm(queries, hWv, bv, Vt, M1, ED, ED, 0);
    {
        dim3 grid((QLEN + 15) / 16, NHD, NB);
        mha_attn_kernel<<<grid, 32, 0, stream>>>(Qt, Kt, Vt, mo, QLEN);
    }
    gemm(mo, hWo, bo, out1, M1, ED, ED, 0);
    add_ln_kernel<<<M1, 256, 0, stream>>>(queries, out1, ln1_g, ln1_b, out2);

    // ---- deformable attention block ----
    add_kernel<<<(M1 * ED / 4 + 255) / 256, 256, 0, stream>>>(out2, origq, dq, M1 * ED / 4);
    gemm(bev, hdv, dv_b, value, MV, ED, ED, 0);           // value projection (big GEMM)
    gemm(dq, hdoff, doff_b, offb, M1, NHD * NPTS * 2, ED, 0);
    gemm(dq, hdatt, dattn_b, attnb, M1, NHD * NPTS, ED, 0);
    {
        dim3 grid(QLEN, NHD, NB);
        deform_sample_kernel<<<grid, 32, 0, stream>>>(value, refpts, offb, attnb, samp);
    }
    gemm(samp, hdout, dout_b, dob, M1, ED, ED, 0);
    add_ln_kernel<<<M1, 256, 0, stream>>>(out2, dob, ln2_g, ln2_b, out3);

    // ---- FFN + final LN ----
    gemm(out3, hW1, b1, ffh, M1, DFF, ED, 1);             // ReLU fused
    gemm(ffh, hW2, b2, out4, M1, ED, DFF, 0);
    add_ln_kernel<<<M1, 256, 0, stream>>>(out3, out4, ln3_g, ln3_b, out);
}
